// SANet_69406671503787
// MI455X (gfx1250) — matrix operand order repaired and checked by reading, not compiled, not run on hardware
//
#include <hip/hip_runtime.h>

// ---------------------------------------------------------------------------
// SANet attention for MI455X (gfx1250, wave32, WMMA bf16 + TDM async tensor)
//   B=16, C=512, H=W=64 (HW=4096), fp32 in/out, bf16 GEMMs internally.
// ---------------------------------------------------------------------------

typedef __attribute__((ext_vector_type(16))) __bf16 v16bf;
typedef __attribute__((ext_vector_type(8)))  __bf16 v8bf;
typedef __attribute__((ext_vector_type(4)))  __bf16 v4bf;
typedef __attribute__((ext_vector_type(8)))  float  v8f;
typedef __attribute__((ext_vector_type(4)))  unsigned int v4u;
typedef __attribute__((ext_vector_type(8)))  int    v8i;
typedef __attribute__((ext_vector_type(4)))  int    v4i;

#define BATCH 16
#define CCH   512
#define HW    4096

#if defined(__has_builtin)
#  if __has_builtin(__builtin_amdgcn_tensor_load_to_lds) && \
      __has_builtin(__builtin_amdgcn_s_wait_tensorcnt)
#    define HAVE_TDM 1
#  else
#    define HAVE_TDM 0
#  endif
#else
#  define HAVE_TDM 0
#endif

// ---------------------------------------------------------------------------
// WMMA helper: v_wmma_f32_16x16x32_bf16, D = A(16x32) x B(32x16) + C
// ---------------------------------------------------------------------------
__device__ __forceinline__ v8f wmma_bf16(v16bf a, v16bf b, v8f c) {
  return __builtin_amdgcn_wmma_f32_16x16x32_bf16(false, a, false, b,
                                                 (short)0, c, false, false);
}

// A fragment (16x32, 16-bit): lane<16 -> row M=lane, K in {0..7,16..23};
// lane>=16 -> row M=lane-16, K in {8..15,24..31}. LDS tile row-major [M][K].
__device__ __forceinline__ v16bf ld_frag_a(const __bf16* base, int stride,
                                           int m0, int lane) {
  const int row = m0 + (lane & 15);
  const int kh  = (lane >> 4) << 3;
  v8bf lo = *(const v8bf*)(base + (size_t)row * stride + kh);
  v8bf hi = *(const v8bf*)(base + (size_t)row * stride + kh + 16);
  v16bf r;
#pragma unroll
  for (int i = 0; i < 8; ++i) { r[i] = lo[i]; r[i + 8] = hi[i]; }
  return r;
}

// B fragment (32x16, 16-bit): lane<16 -> col N=lane, K in {0..7,16..23};
// lane>=16 -> col N=lane-16, K in {8..15,24..31}. LDS tile [N][K] row-major, `stride` elems/row.
__device__ __forceinline__ v16bf ld_frag_b(const __bf16* base, int stride,
                                           int n0, int lane) {
  const int row = n0 + (lane & 15);
  const int kh  = (lane >> 4) << 3;
  v8bf lo = *(const v8bf*)(base + (size_t)row * stride + kh);
  v8bf hi = *(const v8bf*)(base + (size_t)row * stride + kh + 16);
  v16bf r;
#pragma unroll
  for (int i = 0; i < 8; ++i) { r[i] = lo[i]; r[i + 8] = hi[i]; }
  return r;
}

#if HAVE_TDM
// ---------------------------------------------------------------------------
// Tensor Data Mover: 2D bf16 tile load (tile_w x tile_h) from a row-major
// matrix with row stride `stride` (elements) into LDS at byte offset lds_off.
// Optional LDS padding: pad_amount+1 DWORDs every 2<<pad_interval DWORDs.
// ---------------------------------------------------------------------------
__device__ __forceinline__ unsigned lds_off_bytes(const void* p) {
  // Generic LDS addresses carry the LDS byte offset in the low 32 bits.
  return (unsigned)(unsigned long long)p;
}

__device__ __forceinline__ void tdm_load_2d(const void* gsrc, unsigned lds_off,
                                            unsigned tile_w, unsigned tile_h,
                                            unsigned stride,
                                            unsigned pad_interval,
                                            unsigned pad_amount,
                                            bool pad_en) {
  const unsigned long long ga = (unsigned long long)gsrc;
  v4u g0;
  g0[0] = 1u;                                     // count=1 (valid), user mode
  g0[1] = lds_off;                                // lds_addr (bytes)
  g0[2] = (unsigned)(ga & 0xffffffffu);           // global_addr[31:0]
  g0[3] = (unsigned)((ga >> 32) & 0x01ffffffu)    // global_addr[56:32]
          | (2u << 30);                           // type=2 ("image")
  v8i g1;
  g1[0] = (int)((1u << 16)                        // data_size=1 -> 2 bytes
                | (pad_en ? (1u << 20) : 0u)      // pad_enable
                | (pad_interval << 22)            // pad_interval (3 -> 16 DW)
                | (pad_amount << 25));            // pad_amount  (3 -> 4 DW)
  g1[1] = (int)(stride << 16);                    // tensor_dim0[15:0]
  g1[2] = (int)((stride >> 16) | ((unsigned)CCH << 16)); // dim0 hi / dim1 lo
  g1[3] = (int)(tile_w << 16);                    // dim1 hi=0, tile_dim0
  g1[4] = (int)tile_h;                            // tile_dim1, tile_dim2=0
  g1[5] = (int)stride;                            // tensor_dim0_stride[31:0]
  g1[6] = 0;                                      // stride hi, dim1_stride lo
  g1[7] = 0;
  const v4i z4 = {0, 0, 0, 0};
#if __clang_major__ >= 23
  const v8i z8 = {0, 0, 0, 0, 0, 0, 0, 0};
  __builtin_amdgcn_tensor_load_to_lds(g0, g1, z4, z4, z8, 0);
#else
  __builtin_amdgcn_tensor_load_to_lds(g0, g1, z4, z4, 0);
#endif
}
#endif  // HAVE_TDM

// ---------------------------------------------------------------------------
// Kernel 1: instance-norm statistics (mean, rstd) per (b, c) over HW=4096.
// grid = (B*C, 2), 256 threads. blockIdx.y: 0 -> f_c, 1 -> f_s.
// ---------------------------------------------------------------------------
__global__ __launch_bounds__(256) void in_stats_kernel(
    const float* __restrict__ f_c, const float* __restrict__ f_s,
    float* __restrict__ mu_c, float* __restrict__ rs_c,
    float* __restrict__ mu_s, float* __restrict__ rs_s)
{
  const int bc = blockIdx.x;
  const float* __restrict__ src = (blockIdx.y == 0) ? f_c : f_s;
  const float4* __restrict__ p =
      (const float4*)(src + (size_t)bc * HW);

  float s = 0.f, s2 = 0.f;
#pragma unroll
  for (int t = 0; t < 4; ++t) {
    const float4 v = p[threadIdx.x + t * 256];
    s  += v.x + v.y + v.z + v.w;
    s2 += v.x * v.x + v.y * v.y + v.z * v.z + v.w * v.w;
  }
  __shared__ float red0[256];
  __shared__ float red1[256];
  red0[threadIdx.x] = s;
  red1[threadIdx.x] = s2;
  __syncthreads();
#pragma unroll
  for (int off = 128; off > 0; off >>= 1) {
    if (threadIdx.x < off) {
      red0[threadIdx.x] += red0[threadIdx.x + off];
      red1[threadIdx.x] += red1[threadIdx.x + off];
    }
    __syncthreads();
  }
  if (threadIdx.x == 0) {
    float m   = red0[0] * (1.f / HW);
    float var = red1[0] * (1.f / HW) - m * m;
    float rs  = rsqrtf(var + 1e-5f);
    if (blockIdx.y == 0) { mu_c[bc] = m; rs_c[bc] = rs; }
    else                 { mu_s[bc] = m; rs_s[bc] = rs; }
  }
}

// ---------------------------------------------------------------------------
// Kernel 2: fused (normalize) + 1x1 conv GEMM, bf16 output.
//   out[b,o,n] = sum_c W[o,c] * xhat[b,c,n] + bias[o]
// grid = (HW/128, C/128, B*3); z%3: 0->fc(norm f_c,w1) 1->gs(norm f_s,w2)
// 2->hs(raw f_s,w3). 256 thr = 8 wave32, 128x128 tile, BK=32, register-
// prefetch pipeline: next K-tile global loads overlap current WMMAs.
// ---------------------------------------------------------------------------
__global__ __launch_bounds__(256) void conv1x1_gemm_kernel(
    const float* __restrict__ f_c, const float* __restrict__ f_s,
    const float* __restrict__ w1, const float* __restrict__ b1,
    const float* __restrict__ w2, const float* __restrict__ b2,
    const float* __restrict__ w3, const float* __restrict__ b3,
    const float* __restrict__ mu_c, const float* __restrict__ rs_c,
    const float* __restrict__ mu_s, const float* __restrict__ rs_s,
    __bf16* __restrict__ fcb, __bf16* __restrict__ gsb, __bf16* __restrict__ hsb)
{
  __shared__ __bf16 sA[128 * 32];        // W tile [M][K]
  __shared__ __bf16 sB[128 * 40];        // X tile [N][K] (+8 pad)

  const int n0 = blockIdx.x * 128;
  const int m0 = blockIdx.y * 128;
  const int b  = blockIdx.z / 3;
  const int j  = blockIdx.z % 3;

  const float* __restrict__ X   = (j == 0) ? f_c : f_s;
  const float* __restrict__ W   = (j == 0) ? w1 : (j == 1) ? w2 : w3;
  const float* __restrict__ bia = (j == 0) ? b1 : (j == 1) ? b2 : b3;
  const float* __restrict__ mu  = (j == 0) ? mu_c : mu_s;
  const float* __restrict__ rs  = (j == 0) ? rs_c : rs_s;
  const bool norm = (j != 2);
  __bf16* __restrict__ O = (j == 0) ? fcb : (j == 1) ? gsb : hsb;

  X  += (size_t)b * CCH * HW;
  O  += (size_t)b * CCH * HW;
  mu += b * CCH;
  rs += b * CCH;

  const int tid  = threadIdx.x;
  const int lane = tid & 31;
  const int wave = tid >> 5;
  const int wm   = (wave >> 1) * 32;
  const int wn   = (wave & 1) * 64;

  v8f acc[2][4] = {};

  float4 wr[4], xr[4];
  float  muv[4], rsv[4];

  // Vectorized prefetch of one K-tile into registers (coalesced float4).
  auto loadRegs = [&](int k0) {
#pragma unroll
    for (int t = 0; t < 4; ++t) {
      const int iw = tid + t * 256;
      const int r  = iw >> 3, c4 = (iw & 7) << 2;
      wr[t] = *(const float4*)&W[(size_t)(m0 + r) * CCH + k0 + c4];
      const int k = iw >> 5, n4 = (iw & 31) << 2;
      xr[t]  = *(const float4*)&X[(size_t)(k0 + k) * HW + n0 + n4];
      muv[t] = norm ? mu[k0 + k] : 0.f;
      rsv[t] = norm ? rs[k0 + k] : 1.f;
    }
  };

  auto storeLds = [&]() {
#pragma unroll
    for (int t = 0; t < 4; ++t) {
      const int iw = tid + t * 256;
      const int r  = iw >> 3, c4 = (iw & 7) << 2;
      v4bf wv;
      wv[0] = (__bf16)wr[t].x; wv[1] = (__bf16)wr[t].y;
      wv[2] = (__bf16)wr[t].z; wv[3] = (__bf16)wr[t].w;
      *(v4bf*)&sA[r * 32 + c4] = wv;
      const int k = iw >> 5, n4 = (iw & 31) << 2;
      const float xs[4] = {xr[t].x, xr[t].y, xr[t].z, xr[t].w};
#pragma unroll
      for (int e = 0; e < 4; ++e)
        sB[(n4 + e) * 40 + k] = (__bf16)((xs[e] - muv[t]) * rsv[t]);
    }
  };

  loadRegs(0);
#pragma unroll 1
  for (int kt = 0; kt < CCH / 32; ++kt) {
    __syncthreads();                 // previous tile fully consumed
    storeLds();
    __syncthreads();                 // stores visible
    if (kt + 1 < CCH / 32) loadRegs((kt + 1) * 32);  // overlap w/ WMMAs

    const v16bf a0 = ld_frag_a(sA, 32, wm, lane);
    const v16bf a1 = ld_frag_a(sA, 32, wm + 16, lane);
#pragma unroll
    for (int nf = 0; nf < 4; ++nf) {
      const v16bf bfr = ld_frag_b(sB, 40, wn + nf * 16, lane);
      acc[0][nf] = wmma_bf16(a0, bfr, acc[0][nf]);
      acc[1][nf] = wmma_bf16(a1, bfr, acc[1][nf]);
    }
  }

  // Epilogue: C/D layout -> M = r + 8*(lane>=16), N = lane&15. Add bias.
#pragma unroll
  for (int mf = 0; mf < 2; ++mf) {
#pragma unroll
    for (int nf = 0; nf < 4; ++nf) {
      const int mB = m0 + wm + mf * 16 + ((lane >> 4) << 3);
      const int nB = n0 + wn + nf * 16 + (lane & 15);
#pragma unroll
      for (int r = 0; r < 8; ++r) {
        const float v = acc[mf][nf][r] + bia[mB + r];
        O[(size_t)(mB + r) * HW + nB] = (__bf16)v;
      }
    }
  }
}

// ---------------------------------------------------------------------------
// Kernel 3: attn[b] = gs[b] (C x HW) * fc[b]^T -> (C x C), K = HW = 4096.
// Both operands row-major over K -> pure 2D tile copies: drive them with the
// Tensor Data Mover, double-buffered in LDS (wave0 issues, all waves compute).
// grid = (C/128, C/128, B).
// ---------------------------------------------------------------------------
__global__ __launch_bounds__(256) void attn_gemm_kernel(
    const __bf16* __restrict__ gsb, const __bf16* __restrict__ fcb,
    __bf16* __restrict__ attnb)
{
  __shared__ __bf16 sA[2][128 * 32];
  __shared__ __bf16 sB[2][128 * 40];

  const int n0 = blockIdx.x * 128;
  const int m0 = blockIdx.y * 128;
  const int b  = blockIdx.z;
  const __bf16* __restrict__ A  = gsb + (size_t)b * CCH * HW + (size_t)m0 * HW;
  const __bf16* __restrict__ Bm = fcb + (size_t)b * CCH * HW + (size_t)n0 * HW;

  const int tid  = threadIdx.x;
  const int lane = tid & 31;
  const int wave = tid >> 5;
  const int wm   = (wave >> 1) * 32;
  const int wn   = (wave & 1) * 64;

  v8f acc[2][4] = {};

#if HAVE_TDM
  if (wave == 0) {
    tdm_load_2d(A,  lds_off_bytes(&sA[0][0]), 32, 128, HW, 0, 0, false);
    tdm_load_2d(Bm, lds_off_bytes(&sB[0][0]), 32, 128, HW, 3, 3, true);
    __builtin_amdgcn_s_wait_tensorcnt(0);
  }
  __syncthreads();
#pragma unroll 1
  for (int kt = 0; kt < HW / 32; ++kt) {
    const int cur = kt & 1;
    if ((kt + 1 < HW / 32) && wave == 0) {
      const int k1 = (kt + 1) * 32;
      tdm_load_2d(A + k1,  lds_off_bytes(&sA[cur ^ 1][0]), 32, 128, HW, 0, 0, false);
      tdm_load_2d(Bm + k1, lds_off_bytes(&sB[cur ^ 1][0]), 32, 128, HW, 3, 3, true);
    }
    const v16bf a0 = ld_frag_a(sA[cur], 32, wm, lane);
    const v16bf a1 = ld_frag_a(sA[cur], 32, wm + 16, lane);
#pragma unroll
    for (int nf = 0; nf < 4; ++nf) {
      const v16bf bfr = ld_frag_b(sB[cur], 40, wn + nf * 16, lane);
      acc[0][nf] = wmma_bf16(a0, bfr, acc[0][nf]);
      acc[1][nf] = wmma_bf16(a1, bfr, acc[1][nf]);
    }
    if (wave == 0) __builtin_amdgcn_s_wait_tensorcnt(0);
    __syncthreads();
  }
#else
  // Fallback: vectorized VGPR-staged copies, single buffer.
#pragma unroll 1
  for (int kt = 0; kt < HW / 32; ++kt) {
    const int k0 = kt * 32;
#pragma unroll
    for (int t = 0; t < 2; ++t) {
      const int i = tid + t * 256;
      const int r = i >> 2, c8 = (i & 3) << 3;
      *(uint4*)&sA[0][r * 32 + c8] =
          *(const uint4*)&A[(size_t)r * HW + k0 + c8];
      *(uint4*)&sB[0][r * 40 + c8] =
          *(const uint4*)&Bm[(size_t)r * HW + k0 + c8];
    }
    __syncthreads();
    const v16bf a0 = ld_frag_a(sA[0], 32, wm, lane);
    const v16bf a1 = ld_frag_a(sA[0], 32, wm + 16, lane);
#pragma unroll
    for (int nf = 0; nf < 4; ++nf) {
      const v16bf bfr = ld_frag_b(sB[0], 40, wn + nf * 16, lane);
      acc[0][nf] = wmma_bf16(a0, bfr, acc[0][nf]);
      acc[1][nf] = wmma_bf16(a1, bfr, acc[1][nf]);
    }
    __syncthreads();
  }
#endif

  __bf16* __restrict__ Od = attnb + (size_t)b * CCH * CCH;
#pragma unroll
  for (int mf = 0; mf < 2; ++mf) {
#pragma unroll
    for (int nf = 0; nf < 4; ++nf) {
      const int mB = m0 + wm + mf * 16 + ((lane >> 4) << 3);
      const int nB = n0 + wn + nf * 16 + (lane & 15);
#pragma unroll
      for (int r = 0; r < 8; ++r)
        Od[(size_t)(mB + r) * CCH + nB] = (__bf16)acc[mf][nf][r];
    }
  }
}

// ---------------------------------------------------------------------------
// Kernel 4: out[b] = attn[b] (C x C) * hs[b] (C x HW), fp32 epilogue.
// Register-prefetch pipeline; hs tile transposed into LDS [N][K].
// grid = (HW/128, C/128, B).
// ---------------------------------------------------------------------------
__global__ __launch_bounds__(256) void out_gemm_kernel(
    const __bf16* __restrict__ attnb, const __bf16* __restrict__ hsb,
    float* __restrict__ out)
{
  __shared__ __bf16 sA[128 * 32];
  __shared__ __bf16 sB[128 * 40];

  const int n0 = blockIdx.x * 128;
  const int m0 = blockIdx.y * 128;
  const int b  = blockIdx.z;
  const __bf16* __restrict__ A = attnb + (size_t)b * CCH * CCH;
  const __bf16* __restrict__ H = hsb + (size_t)b * CCH * HW;
  float* __restrict__ O = out + (size_t)b * CCH * HW;

  const int tid  = threadIdx.x;
  const int lane = tid & 31;
  const int wave = tid >> 5;
  const int wm   = (wave >> 1) * 32;
  const int wn   = (wave & 1) * 64;

  v8f acc[2][4] = {};

  uint4 ar[2];
  v8bf  hr[2];

  auto loadRegs = [&](int k0) {
#pragma unroll
    for (int t = 0; t < 2; ++t) {
      const int i = tid + t * 256;
      const int r = i >> 2, c8 = (i & 3) << 3;
      ar[t] = *(const uint4*)&A[(size_t)(m0 + r) * CCH + k0 + c8];
      const int k = i >> 4, n8 = (i & 15) << 3;
      hr[t] = *(const v8bf*)&H[(size_t)(k0 + k) * HW + n0 + n8];
    }
  };

  auto storeLds = [&]() {
#pragma unroll
    for (int t = 0; t < 2; ++t) {
      const int i = tid + t * 256;
      const int r = i >> 2, c8 = (i & 3) << 3;
      *(uint4*)&sA[r * 32 + c8] = ar[t];
      const int k = i >> 4, n8 = (i & 15) << 3;
#pragma unroll
      for (int e = 0; e < 8; ++e)
        sB[(n8 + e) * 40 + k] = hr[t][e];
    }
  };

  loadRegs(0);
#pragma unroll 1
  for (int kt = 0; kt < CCH / 32; ++kt) {
    __syncthreads();
    storeLds();
    __syncthreads();
    if (kt + 1 < CCH / 32) loadRegs((kt + 1) * 32);

    const v16bf a0 = ld_frag_a(sA, 32, wm, lane);
    const v16bf a1 = ld_frag_a(sA, 32, wm + 16, lane);
#pragma unroll
    for (int nf = 0; nf < 4; ++nf) {
      const v16bf bfr = ld_frag_b(sB, 40, wn + nf * 16, lane);
      acc[0][nf] = wmma_bf16(a0, bfr, acc[0][nf]);
      acc[1][nf] = wmma_bf16(a1, bfr, acc[1][nf]);
    }
  }

#pragma unroll
  for (int mf = 0; mf < 2; ++mf) {
#pragma unroll
    for (int nf = 0; nf < 4; ++nf) {
      const int mB = m0 + wm + mf * 16 + ((lane >> 4) << 3);
      const int nB = n0 + wn + nf * 16 + (lane & 15);
#pragma unroll
      for (int r = 0; r < 8; ++r)
        O[(size_t)(mB + r) * HW + nB] = acc[mf][nf][r];
    }
  }
}

// ---------------------------------------------------------------------------
// Host launcher
// ---------------------------------------------------------------------------
extern "C" void kernel_launch(void* const* d_in, const int* in_sizes, int n_in,
                              void* d_out, int out_size, void* d_ws, size_t ws_size,
                              hipStream_t stream) {
  const float* f_c = (const float*)d_in[0];
  const float* f_s = (const float*)d_in[1];
  const float* w1  = (const float*)d_in[2];
  const float* b1  = (const float*)d_in[3];
  const float* w2  = (const float*)d_in[4];
  const float* b2  = (const float*)d_in[5];
  const float* w3  = (const float*)d_in[6];
  const float* b3  = (const float*)d_in[7];

  // Workspace: fp32 stats, then bf16 fc/gs/hs (B,C,HW), bf16 attn (B,C,C).
  float* mu_c = (float*)d_ws;
  float* rs_c = mu_c + BATCH * CCH;
  float* mu_s = rs_c + BATCH * CCH;
  float* rs_s = mu_s + BATCH * CCH;
  __bf16* fcb = (__bf16*)(rs_s + BATCH * CCH);
  const size_t PER = (size_t)BATCH * CCH * HW;
  __bf16* gsb   = fcb + PER;
  __bf16* hsb   = gsb + PER;
  __bf16* attnb = hsb + PER;

  in_stats_kernel<<<dim3(BATCH * CCH, 2), 256, 0, stream>>>(
      f_c, f_s, mu_c, rs_c, mu_s, rs_s);

  conv1x1_gemm_kernel<<<dim3(HW / 128, CCH / 128, BATCH * 3), 256, 0, stream>>>(
      f_c, f_s, w1, b1, w2, b2, w3, b3,
      mu_c, rs_c, mu_s, rs_s, fcb, gsb, hsb);

  attn_gemm_kernel<<<dim3(CCH / 128, CCH / 128, BATCH), 256, 0, stream>>>(
      gsb, fcb, attnb);

  out_gemm_kernel<<<dim3(HW / 128, CCH / 128, BATCH), 256, 0, stream>>>(
      attnb, hsb, (float*)d_out);
}
